// GraphTransformer_55972013802259
// MI455X (gfx1250) — matrix operand order repaired and checked by reading, not compiled, not run on hardware
//
#include <hip/hip_runtime.h>

typedef __attribute__((ext_vector_type(16))) _Float16 v16h;
typedef __attribute__((ext_vector_type(8)))  float    v8f;

#define POSD 3
#define FDIM 128
#define MDIM 16
#define FE   16
#define XROW (POSD + FDIM)      // 131
#define EIN  289
#define EIN_PAD 320
#define EHID 578
#define EHID_PAD 608
#define NIN  144
#define NIN_PAD 160
#define NHID 256
#define KT1  10   // EIN_PAD/32
#define NT1  38   // EHID_PAD/16
#define KT2  19   // EHID_PAD/32
#define KT3  5    // NIN_PAD/32
#define NT3  16   // NHID/16
#define KT4  8    // NHID/32
#define NT4  8    // FDIM/16
#define LNEPS 1e-5f

#define TME 64            // edges per block (4 M-tiles, one per wave)
#define TMN 64            // nodes per block

// LDS row strides (halves / floats) padded to avoid 64-bank conflicts
#define SIN_STR   328     // >= EIN_PAD
#define SHID_STR  616     // >= EHID_PAD
#define NIN_STR   168     // >= NIN_PAD
#define NHID_STR  264     // >= NHID
#define SO_STR    132     // >= FDIM (floats)

// edge-kernel dynamic LDS layout (bytes)
#define E_SIN_B   (TME * SIN_STR * 2)            // 41984
#define E_SHID_B  (TME * SHID_STR * 2)           // 78848
#define E_SOUT_B  (TME * 16 * 4)                 // 4096
#define E_SDST_B  (TME * 4)                      // 256
#define E_SMEM_B  (E_SIN_B + E_SHID_B + E_SOUT_B + E_SDST_B)

// node-kernel dynamic LDS layout (bytes)
#define N_SIN_B   (TMN * NIN_STR * 2)            // 21504
#define N_SHID_B  (TMN * NHID_STR * 2)           // 33792
#define N_SO_B    (TMN * SO_STR * 4)             // 33792
#define N_SMEM_B  (N_SIN_B + N_SHID_B + N_SO_B)

__device__ __forceinline__ float silu_f(float v) { return v / (1.f + __expf(-v)); }

// Load a 16x32 f16 A-tile from an LDS row (row-major, even col alignment) into
// the CDNA5 WMMA A layout: lanes 0-15 -> K base 0, lanes 16-31 -> K base 8;
// VGPR r (elements 2r,2r+1): K = base + (r<4 ? 2r : 16+2(r-4)) (+ kt*32).
__device__ __forceinline__ v16h load_a_tile(const _Float16* rowp, int kt, int grp) {
  union { v16h v; unsigned u[8]; } t;
#pragma unroll
  for (int r = 0; r < 8; ++r) {
    const int kb = kt * 32 + grp * 8 + ((r < 4) ? (r * 2) : (16 + (r - 4) * 2));
    t.u[r] = *(const unsigned*)(rowp + kb);
  }
  return t.v;
}

// ---------------------------------------------------------------------------
// Weight packing: f32 [K,Ncols] row-major -> f16 B-tile layout.
// dst[((nt*KT + kt)*32 + lane)*16 + e] = w[k, n], k = kt*32 + (e<8 ? (lane/16)*8 + e : 16 + (lane/16)*8 + e-8),
// n = nt*16 + lane%16; zero-padded out of range.
// ---------------------------------------------------------------------------
__global__ void pack_kernel(const float* __restrict__ w, _Float16* __restrict__ dst,
                            int K, int Ncols, int KT, int NT) {
  const int idx = blockIdx.x * 256 + threadIdx.x;
  const int total = NT * KT * 512;
  if (idx >= total) return;
  const int e    = idx & 15;
  const int lane = (idx >> 4) & 31;
  const int kt   = (idx >> 9) % KT;
  const int nt   = idx / (KT * 512);
  const int k = kt * 32 + ((e < 8) ? ((lane >> 4) * 8 + e) : (16 + (lane >> 4) * 8 + (e - 8)));
  const int n = nt * 16 + (lane & 15);
  const float v = (k < K && n < Ncols) ? w[(size_t)k * Ncols + n] : 0.f;
  dst[idx] = (_Float16)v;
}

__global__ void zero_kernel(float* __restrict__ p, int n) {
  const int i = blockIdx.x * 256 + threadIdx.x;
  if (i < n) p[i] = 0.f;
}

// ---------------------------------------------------------------------------
// Edge kernel: 64 edges per block, 128 threads (4 waves, wave = M-tile).
// All waves walk the same N-tile sequence so B loads are shared in WGP$.
// ---------------------------------------------------------------------------
__global__ __launch_bounds__(128) void edge_kernel(
    const float* __restrict__ x, const long long* __restrict__ ei,
    const _Float16* __restrict__ w1p, const float* __restrict__ b1,
    const _Float16* __restrict__ w2p, const float* __restrict__ b2,
    const float* __restrict__ g1, const float* __restrict__ gb1,
    float* __restrict__ macc, int E_) {
  extern __shared__ char smem[];
  _Float16* sIn  = (_Float16*)smem;                                 // [TME][SIN_STR]
  _Float16* sHid = (_Float16*)(smem + E_SIN_B);                     // [TME][SHID_STR]
  float*    sOut = (float*)   (smem + E_SIN_B + E_SHID_B);          // [TME][16]
  int*      sDst = (int*)     (smem + E_SIN_B + E_SHID_B + E_SOUT_B);

  const int tid = threadIdx.x;
  const int e0  = blockIdx.x * TME;

  {
    const int row  = tid >> 1;        // 0..63
    const int sub  = tid & 1;
    const int edge = e0 + row;
    const bool ev  = edge < E_;
    const long long si = ev ? ei[edge] : 0;
    const long long di = ev ? ei[E_ + edge] : 0;
    if (sub == 0) sDst[row] = ev ? (int)di : -1;
    const float* xs = x + (size_t)si * XROW;
    const float* xd = x + (size_t)di * XROW;
    const float rx = xs[0] - xd[0], ry = xs[1] - xd[1], rz = xs[2] - xd[2];
    const float d = rx * rx + ry * ry + rz * rz;
    _Float16* in = sIn + (size_t)row * SIN_STR;
    for (int j = sub; j < FDIM; j += 2) {
      in[j]        = (_Float16)(ev ? xd[POSD + j] : 0.f);
      in[FDIM + j] = (_Float16)(ev ? xs[POSD + j] : 0.f);
    }
    for (int j = sub; j < FE; j += 2) {
      const float ds = ldexpf(d, -j);
      in[2 * FDIM + j]      = (_Float16)(ev ? __sinf(ds) : 0.f);
      in[2 * FDIM + FE + j] = (_Float16)(ev ? __cosf(ds) : 0.f);
    }
    if (sub == 0) in[2 * FDIM + 2 * FE] = (_Float16)(ev ? d : 0.f);
    for (int j = EIN + sub; j < EIN_PAD; j += 2) in[j] = (_Float16)0.f;
  }
  __syncthreads();

  const int wave = tid >> 5;
  const int lane = tid & 31;
  const int mrow = lane & 15;
  const int grp  = lane >> 4;

  const _Float16* inrow = sIn + (size_t)(wave * 16 + mrow) * SIN_STR;
  v16h a[KT1];
#pragma unroll
  for (int kt = 0; kt < KT1; ++kt) a[kt] = load_a_tile(inrow, kt, grp);

  // Layer 1: [64,320] x [320,608]; each wave does its own M-tile, same nt order.
  for (int nt = 0; nt < NT1; ++nt) {
    const int n = nt * 16 + mrow;
    const float bias = (n < EHID) ? b1[n] : 0.f;
    v8f acc = {};
    const _Float16* bp = w1p + ((size_t)nt * KT1 * 32 + lane) * 16;
#pragma unroll
    for (int kt = 0; kt < KT1; ++kt) {
      v16h b = *(const v16h*)(bp + (size_t)kt * 512);
      acc = __builtin_amdgcn_wmma_f32_16x16x32_f16(false, a[kt], false, b,
                                                   (short)0, acc, false, false);
    }
#pragma unroll
    for (int r = 0; r < 8; ++r)
      sHid[(size_t)(wave * 16 + r + grp * 8) * SHID_STR + n] =
          (_Float16)silu_f(acc[r] + bias);
  }
  __syncthreads();

  // Layer 2: [64,608] x [608,16]; one M-tile per wave.
  {
    const _Float16* hrow = sHid + (size_t)(wave * 16 + mrow) * SHID_STR;
    v8f acc = {};
    for (int kt = 0; kt < KT2; ++kt) {
      v16h a2 = load_a_tile(hrow, kt, grp);
      v16h b  = *(const v16h*)(w2p + ((size_t)kt * 32 + lane) * 16);
      acc = __builtin_amdgcn_wmma_f32_16x16x32_f16(false, a2, false, b,
                                                   (short)0, acc, false, false);
    }
    const float bias = b2[mrow];
#pragma unroll
    for (int r = 0; r < 8; ++r)
      sOut[(size_t)(wave * 16 + r + grp * 8) * 16 + mrow] = silu_f(acc[r] + bias);
  }
  __syncthreads();

  // LayerNorm(16) + scatter-add into node accumulator.
  if (tid < TME) {
    const int dst = sDst[tid];
    if (dst >= 0) {
      const float* o = sOut + (size_t)tid * 16;
      float mean = 0.f;
#pragma unroll
      for (int j = 0; j < 16; ++j) mean += o[j];
      mean *= (1.f / 16.f);
      float var = 0.f;
#pragma unroll
      for (int j = 0; j < 16; ++j) { const float t = o[j] - mean; var += t * t; }
      var *= (1.f / 16.f);
      const float rs = rsqrtf(var + LNEPS);
      float* outp = macc + (size_t)dst * MDIM;
#pragma unroll
      for (int j = 0; j < 16; ++j)
        unsafeAtomicAdd(outp + j, (o[j] - mean) * rs * g1[j] + gb1[j]);
    }
  }
}

// ---------------------------------------------------------------------------
// Node kernel: 64 nodes per block, 128 threads (4 waves, wave = M-tile).
// ---------------------------------------------------------------------------
__global__ __launch_bounds__(128) void node_kernel(
    const float* __restrict__ x, const float* __restrict__ macc,
    const float* __restrict__ en2g, const float* __restrict__ en2b,
    const float* __restrict__ nn1g, const float* __restrict__ nn1b,
    const _Float16* __restrict__ n1p, const float* __restrict__ nb1,
    const _Float16* __restrict__ n2p, const float* __restrict__ nb2,
    const float* __restrict__ nn2g, const float* __restrict__ nn2b,
    float* __restrict__ out, int N_) {
  extern __shared__ char smem[];
  _Float16* sIn  = (_Float16*)smem;                       // [TMN][NIN_STR]
  _Float16* sHid = (_Float16*)(smem + N_SIN_B);           // [TMN][NHID_STR]
  float*    sO   = (float*)   (smem + N_SIN_B + N_SHID_B);// [TMN][SO_STR]

  const int tid  = threadIdx.x;
  const int n0   = blockIdx.x * TMN;
  const int row  = tid >> 1;          // 0..63
  const int sub  = tid & 1;
  const int node = n0 + row;
  const bool nv  = node < N_;
  const float* xr = x + (size_t)node * XROW;
  _Float16* in = sIn + (size_t)row * NIN_STR;

  {
    // LN(feats) over 128 -> sIn[:,0:128]  (2-thread shuffle reduction per row)
    float s = 0.f, ss = 0.f;
    if (nv)
      for (int j = sub; j < FDIM; j += 2) { const float v = xr[POSD + j]; s += v; ss += v * v; }
    s += __shfl_xor(s, 1, 32); ss += __shfl_xor(ss, 1, 32);
    const float mean = s * (1.f / FDIM);
    const float var  = ss * (1.f / FDIM) - mean * mean;
    const float rs   = rsqrtf(var + LNEPS);
    for (int j = sub; j < FDIM; j += 2)
      in[j] = (_Float16)(nv ? ((xr[POSD + j] - mean) * rs * nn1g[j] + nn1b[j]) : 0.f);

    // LN(m_i) over 16 -> sIn[:,128:144]; pad 144..159 = 0
    float v[8];
    float s2 = 0.f, ss2 = 0.f;
#pragma unroll
    for (int t = 0; t < 8; ++t) {
      v[t] = nv ? macc[(size_t)node * MDIM + sub * 8 + t] : 0.f;
      s2 += v[t]; ss2 += v[t] * v[t];
    }
    s2 += __shfl_xor(s2, 1, 32); ss2 += __shfl_xor(ss2, 1, 32);
    const float mean2 = s2 * (1.f / 16.f);
    const float var2  = ss2 * (1.f / 16.f) - mean2 * mean2;
    const float rs2   = rsqrtf(var2 + LNEPS);
#pragma unroll
    for (int t = 0; t < 8; ++t) {
      const int j = sub * 8 + t;
      in[FDIM + j] = (_Float16)(nv ? ((v[t] - mean2) * rs2 * en2g[j] + en2b[j]) : 0.f);
    }
    for (int j = NIN + sub; j < NIN_PAD; j += 2) in[j] = (_Float16)0.f;
  }
  __syncthreads();

  const int wave = tid >> 5;
  const int lane = tid & 31;
  const int mrow = lane & 15;
  const int grp  = lane >> 4;

  const _Float16* inrow = sIn + (size_t)(wave * 16 + mrow) * NIN_STR;
  v16h a[KT3];
#pragma unroll
  for (int kt = 0; kt < KT3; ++kt) a[kt] = load_a_tile(inrow, kt, grp);

  // Layer 1: [64,160] x [160,256]
  for (int nt = 0; nt < NT3; ++nt) {
    const int n = nt * 16 + mrow;
    const float bias = nb1[n];
    v8f acc = {};
    const _Float16* bp = n1p + ((size_t)nt * KT3 * 32 + lane) * 16;
#pragma unroll
    for (int kt = 0; kt < KT3; ++kt) {
      v16h b = *(const v16h*)(bp + (size_t)kt * 512);
      acc = __builtin_amdgcn_wmma_f32_16x16x32_f16(false, a[kt], false, b,
                                                   (short)0, acc, false, false);
    }
#pragma unroll
    for (int r = 0; r < 8; ++r)
      sHid[(size_t)(wave * 16 + r + grp * 8) * NHID_STR + n] =
          (_Float16)silu_f(acc[r] + bias);
  }
  __syncthreads();

  // Layer 2: [64,256] x [256,128] (no activation)
  const _Float16* hrow = sHid + (size_t)(wave * 16 + mrow) * NHID_STR;
  for (int nt = 0; nt < NT4; ++nt) {
    const int n = nt * 16 + mrow;
    const float bias = nb2[n];
    v8f acc = {};
    const _Float16* bp = n2p + ((size_t)nt * KT4 * 32 + lane) * 16;
#pragma unroll
    for (int kt = 0; kt < KT4; ++kt) {
      v16h a2 = load_a_tile(hrow, kt, grp);
      v16h b  = *(const v16h*)(bp + (size_t)kt * 512);
      acc = __builtin_amdgcn_wmma_f32_16x16x32_f16(false, a2, false, b,
                                                   (short)0, acc, false, false);
    }
#pragma unroll
    for (int r = 0; r < 8; ++r)
      sO[(size_t)(wave * 16 + r + grp * 8) * SO_STR + n] = acc[r] + bias;
  }
  __syncthreads();

  // Final LN(128) + residual, write [coors, feats+h].
  {
    const float* orow = sO + (size_t)row * SO_STR;
    float s = 0.f, ss = 0.f;
    for (int j = sub; j < FDIM; j += 2) { const float v = orow[j]; s += v; ss += v * v; }
    s += __shfl_xor(s, 1, 32); ss += __shfl_xor(ss, 1, 32);
    const float mean = s * (1.f / FDIM);
    const float var  = ss * (1.f / FDIM) - mean * mean;
    const float rs   = rsqrtf(var + LNEPS);
    if (nv) {
      float* op = out + (size_t)node * XROW;
      if (sub == 0) { op[0] = xr[0]; op[1] = xr[1]; op[2] = xr[2]; }
      for (int j = sub; j < FDIM; j += 2)
        op[POSD + j] = xr[POSD + j] + (orow[j] - mean) * rs * nn2g[j] + nn2b[j];
    }
  }
}

// ---------------------------------------------------------------------------
extern "C" void kernel_launch(void* const* d_in, const int* in_sizes, int n_in,
                              void* d_out, int out_size, void* d_ws, size_t ws_size,
                              hipStream_t stream) {
  const float*     x    = (const float*)d_in[0];
  const long long* ei   = (const long long*)d_in[1];
  const float*     e_w1 = (const float*)d_in[2];
  const float*     e_b1 = (const float*)d_in[3];
  const float*     e_w2 = (const float*)d_in[4];
  const float*     e_b2 = (const float*)d_in[5];
  const float*     en1g = (const float*)d_in[6];
  const float*     en1b = (const float*)d_in[7];
  const float*     en2g = (const float*)d_in[8];
  const float*     en2b = (const float*)d_in[9];
  const float*     nn1g = (const float*)d_in[10];
  const float*     nn1b = (const float*)d_in[11];
  const float*     n_w1 = (const float*)d_in[12];
  const float*     n_b1 = (const float*)d_in[13];
  const float*     n_w2 = (const float*)d_in[14];
  const float*     n_b2 = (const float*)d_in[15];
  const float*     nn2g = (const float*)d_in[16];
  const float*     nn2b = (const float*)d_in[17];

  const int N_ = in_sizes[0] / XROW;
  const int E_ = in_sizes[1] / 2;

  char* ws = (char*)d_ws;
  size_t off = 0;
  float* macc = (float*)(ws + off);
  off += (size_t)N_ * MDIM * sizeof(float);       off = (off + 255) & ~(size_t)255;
  _Float16* w1p = (_Float16*)(ws + off);
  off += (size_t)NT1 * KT1 * 512 * 2;             off = (off + 255) & ~(size_t)255;
  _Float16* w2p = (_Float16*)(ws + off);
  off += (size_t)KT2 * 512 * 2;                   off = (off + 255) & ~(size_t)255;
  _Float16* n1p = (_Float16*)(ws + off);
  off += (size_t)NT3 * KT3 * 512 * 2;             off = (off + 255) & ~(size_t)255;
  _Float16* n2p = (_Float16*)(ws + off);

  zero_kernel<<<(N_ * MDIM + 255) / 256, 256, 0, stream>>>(macc, N_ * MDIM);
  pack_kernel<<<(NT1 * KT1 * 512 + 255) / 256, 256, 0, stream>>>(e_w1, w1p, EIN,  EHID, KT1, NT1);
  pack_kernel<<<(KT2 * 512 + 255) / 256,       256, 0, stream>>>(e_w2, w2p, EHID, MDIM, KT2, 1);
  pack_kernel<<<(NT3 * KT3 * 512 + 255) / 256, 256, 0, stream>>>(n_w1, n1p, NIN,  NHID, KT3, NT3);
  pack_kernel<<<(NT4 * KT4 * 512 + 255) / 256, 256, 0, stream>>>(n_w2, n2p, NHID, FDIM, KT4, NT4);

  edge_kernel<<<(E_ + TME - 1) / TME, 128, E_SMEM_B, stream>>>(
      x, ei, w1p, e_b1, w2p, e_b2, en1g, en1b, macc, E_);
  node_kernel<<<(N_ + TMN - 1) / TMN, 128, N_SMEM_B, stream>>>(
      x, macc, en2g, en2b, nn1g, nn1b, n1p, n_b1, n2p, n_b2, nn2g, nn2b,
      (float*)d_out, N_);
}
